// SingleHeadAttention_66056597013119
// MI455X (gfx1250) — compile-verified, matrix operand order checked by reading, not run on hardware
//
#include <hip/hip_runtime.h>
#include <math.h>

typedef __attribute__((ext_vector_type(16))) __bf16 v16bf;
typedef __attribute__((ext_vector_type(8)))  __bf16 v8bf;
typedef __attribute__((ext_vector_type(8)))  float  v8f;

#define D_MODEL 512
#define BATCH   4
#define SEQ     4096
#define BT      (BATCH*SEQ)   /* 16384 */

/* workspace byte offsets */
#define OFF_XB   ((size_t)0)
#define OFF_QB   ((size_t)16777216)
#define OFF_KB   ((size_t)33554432)
#define OFF_VT   ((size_t)50331648)   /* V stored transposed [D][BT] */
#define OFF_CTX  ((size_t)67108864)
#define OFF_WQ   ((size_t)83886080)
#define OFF_WK   ((size_t)84410368)
#define OFF_WV   ((size_t)84934656)
#define OFF_WO   ((size_t)85458944)

/* attention LDS layout (bytes): double-buffered K/V + per-wave O accum */
#define SM_KV(buf)  ((unsigned)(buf) * 65536u)          /* Kp 32KB + Vp 32KB */
#define SM_VOFF     32768u
#define SM_C        131072   /* + wave*32768 : 16x512 f32 per wave */
#define SM_P        262144   /* + wave*1024  : 16x32 bf16 per wave */
#define SM_L        266240   /* + wave*64    : 16 f32 per wave     */
#define SM_TOTAL    266496

__device__ __forceinline__ v8f wmma_bf16(v16bf a, v16bf b, v8f c) {
  /* (neg_a, A, neg_b, B, c_mod, C, reuse_a, reuse_b) */
  return __builtin_amdgcn_wmma_f32_16x16x32_bf16(false, a, false, b, (short)0, c,
                                                 false, false);
}

/* CDNA5 async copy: 16 global bytes -> 16 LDS bytes, tracked on ASYNCcnt */
__device__ __forceinline__ void async_b128(unsigned lds_byte_off, const void* gaddr) {
  asm volatile("global_load_async_to_lds_b128 %0, %1, off"
               :: "v"(lds_byte_off), "v"(gaddr) : "memory");
}
__device__ __forceinline__ void wait_async0() {
  asm volatile("s_wait_asynccnt 0" ::: "memory");
}

/* ---------------- fp32 -> bf16 cast ---------------- */
__global__ void k_cast_bf16(const float* __restrict__ src,
                            __bf16* __restrict__ dst, int n) {
  int stride = gridDim.x * blockDim.x;
  for (int i = blockIdx.x * blockDim.x + threadIdx.x; i < n; i += stride)
    dst[i] = (__bf16)src[i];
}

/* ---- pack W [E,D] into WMMA B-tile order: tiles (kt of 32, nt of 16),
   512 elements per tile laid out lane-major exactly as the B operand VGPRs
   expect: lane = n + 16*k_hi, element h -> k = k_hi*8 + h (+8 if h>=8). ---- */
__global__ void k_pack_w(const float* __restrict__ W, __bf16* __restrict__ P) {
  int p    = blockIdx.x * 256 + threadIdx.x;   /* 262144 total */
  int tile = p >> 9;
  int w    = p & 511;
  int kt   = tile >> 5;        /* D/32 = 16 */
  int nt   = tile & 31;        /* D/16 = 32 */
  int lane = w >> 4, h = w & 15;
  int n    = lane & 15, lhi = lane >> 4;
  int k    = lhi * 8 + h + ((h & 8) ? 8 : 0);
  int e    = nt * 16 + n;
  int d    = kt * 32 + k;
  P[p] = (__bf16)W[e * D_MODEL + d];
}

/* ------- QKV GEMM: one wave per 64x16 output block (4 row tiles),
   B operands loaded once per k-step and reused 4x. V written transposed.
   launch_bounds(128,1): full VGPR file so 12 accumulators never spill. --- */
__global__ void __launch_bounds__(128, 1)
k_gemm_qkv(const __bf16* __restrict__ xb,
           const __bf16* __restrict__ Wq,
           const __bf16* __restrict__ Wk,
           const __bf16* __restrict__ Wv,
           __bf16* __restrict__ Q,
           __bf16* __restrict__ K,
           __bf16* __restrict__ Vt) {
  int waveId = blockIdx.x * 4 + (threadIdx.x >> 5);  /* 8192 waves */
  int lane   = threadIdx.x & 31;
  int rg     = waveId >> 5;          /* 256 groups of 64 rows */
  int tcol   = waveId & 31;          /* D/16 = 32 col tiles   */
  int m0 = rg * 64, e0 = tcol * 16;
  int row = lane & 15, hi = lane >> 4;

  v8f z = {};
  v8f aq[4], ak[4], av[4];
#pragma unroll
  for (int rt = 0; rt < 4; ++rt) { aq[rt] = z; ak[rt] = z; av[rt] = z; }

  for (int dt = 0; dt < 16; ++dt) {
    size_t boff = ((size_t)(dt * 32 + tcol)) * 512 + lane * 16;
    v16bf bq = *(const v16bf*)(Wq + boff);
    v16bf bk = *(const v16bf*)(Wk + boff);
    v16bf bv = *(const v16bf*)(Wv + boff);
#pragma unroll
    for (int rt = 0; rt < 4; ++rt) {
      const __bf16* ar = xb + (size_t)(m0 + rt * 16 + row) * D_MODEL + hi * 8 + dt * 32;
      v16bf a;
      ((int4*)&a)[0] = *(const int4*)(ar);
      ((int4*)&a)[1] = *(const int4*)(ar + 16);
      aq[rt] = wmma_bf16(a, bq, aq[rt]);
      ak[rt] = wmma_bf16(a, bk, ak[rt]);
      av[rt] = wmma_bf16(a, bv, av[rt]);
    }
  }
#pragma unroll
  for (int rt = 0; rt < 4; ++rt) {
    v8bf vv;
#pragma unroll
    for (int r = 0; r < 8; ++r) {
      int m = rt * 16 + r + hi * 8;
      size_t o = (size_t)(m0 + m) * D_MODEL + e0 + row;
      Q[o] = (__bf16)aq[rt][r];
      K[o] = (__bf16)ak[rt][r];
      vv[r] = (__bf16)av[rt][r];
    }
    /* transposed V: Vt[d][token], 8 consecutive tokens -> one 16B store */
    *(v8bf*)(Vt + (size_t)(e0 + row) * BT + m0 + rt * 16 + hi * 8) = vv;
  }
}

/* ---------------- output projection GEMM (f32 result) ---------------- */
__global__ void __launch_bounds__(128, 1)
k_gemm_out(const __bf16* __restrict__ xb,
           const __bf16* __restrict__ Wp,
           float* __restrict__ out) {
  int waveId = blockIdx.x * 4 + (threadIdx.x >> 5);
  int lane   = threadIdx.x & 31;
  int rg = waveId >> 5, tcol = waveId & 31;
  int m0 = rg * 64, e0 = tcol * 16;
  int row = lane & 15, hi = lane >> 4;

  v8f z = {};
  v8f acc[4];
#pragma unroll
  for (int rt = 0; rt < 4; ++rt) acc[rt] = z;

  for (int dt = 0; dt < 16; ++dt) {
    v16bf b = *(const v16bf*)(Wp + ((size_t)(dt * 32 + tcol)) * 512 + lane * 16);
#pragma unroll
    for (int rt = 0; rt < 4; ++rt) {
      const __bf16* ar = xb + (size_t)(m0 + rt * 16 + row) * D_MODEL + hi * 8 + dt * 32;
      v16bf a;
      ((int4*)&a)[0] = *(const int4*)(ar);
      ((int4*)&a)[1] = *(const int4*)(ar + 16);
      acc[rt] = wmma_bf16(a, b, acc[rt]);
    }
  }
#pragma unroll
  for (int rt = 0; rt < 4; ++rt)
#pragma unroll
    for (int r = 0; r < 8; ++r) {
      int m = rt * 16 + r + hi * 8;
      out[(size_t)(m0 + m) * D_MODEL + e0 + row] = acc[rt][r];
    }
}

/* async-stage one 32-key block of K and V into packed WMMA-B layout.
   Each 16B LDS chunk of the packed layout equals 16 contiguous global
   bytes (K row-major; V transposed), so staging is pure b128 DMA.      */
__device__ __forceinline__ void stage_kv(const __bf16* __restrict__ Kb,
                                         const __bf16* __restrict__ Vt,
                                         int tb, int j0, unsigned smbase,
                                         int tid) {
  for (int c = tid; c < 2048; c += 128) {
    int run  = c & 1;            /* h-run 0..7 / 8..15       */
    int ls   = (c >> 1) & 31;    /* lane slot in packed tile */
    int tile = c >> 6;           /* 0..31                    */
    int n    = ls & 15, khi = ls >> 4;
    int dt = tile >> 1, jn = tile & 1;
    const __bf16* gk = Kb + (size_t)(tb + j0 + jn * 16 + n) * D_MODEL
                          + dt * 32 + khi * 8 + run * 16;
    async_b128(smbase + (unsigned)((tile * 512 + ls * 16) * 2 + run * 16), gk);
    const __bf16* gv = Vt + (size_t)(tile * 16 + n) * BT
                          + tb + j0 + khi * 8 + run * 16;
    async_b128(smbase + SM_VOFF + (unsigned)((tile * 512 + ls * 16) * 2 + run * 16), gv);
  }
}

/* ---------------- flash attention: WG = 4 waves, 64 query rows --------
   Double-buffered async DMA staging of K/V overlapped with WMMA.        */
__global__ void __launch_bounds__(128, 1)
k_attn(const __bf16* __restrict__ Qb,
       const __bf16* __restrict__ Kb,
       const __bf16* __restrict__ Vt,
       __bf16* __restrict__ ctx) {
  extern __shared__ char smem[];
  int wave = threadIdx.x >> 5, lane = threadIdx.x & 31;
  float*  Cw = (float*)(smem + SM_C + (size_t)wave * 32768);
  __bf16* Pw = (__bf16*)(smem + SM_P + (size_t)wave * 1024);
  float*  Lw = (float*)(smem + SM_L + (size_t)wave * 64);

  int b   = blockIdx.y;
  int q0  = blockIdx.x * 64;        /* query base within batch */
  int q0w = q0 + wave * 16;         /* this wave's 16 query rows */
  int row = lane & 15, hi = lane >> 4;
  const int tb = b * SEQ;
  const float scale = 0.044194173824159216f;

  for (int e = lane; e < 16 * D_MODEL; e += 32) Cw[e] = 0.0f;

  float m_i[8], l_i[8];
#pragma unroll
  for (int r = 0; r < 8; ++r) { m_i[r] = -__builtin_inff(); l_i[r] = 0.0f; }

  const __bf16* qrow = Qb + (size_t)(tb + q0w + row) * D_MODEL + hi * 8;

  /* prologue: DMA first key block into buffer 0 */
  stage_kv(Kb, Vt, tb, 0, SM_KV(0), threadIdx.x);

  int buf = 0;
  for (int j0 = 0; j0 <= q0 + 63; j0 += 32) {
    wait_async0();       /* this wave's DMA for current buffer complete */
    __syncthreads();     /* all waves staged + done reading other buf   */

    /* overlap: issue DMA for the next key block while we compute */
    if (j0 + 32 <= q0 + 63)
      stage_kv(Kb, Vt, tb, j0 + 32, SM_KV(buf ^ 1), threadIdx.x);

    if (j0 <= q0w + 15) {           /* not fully above causal diagonal */
      const __bf16* Kp = (const __bf16*)(smem + SM_KV(buf));
      const __bf16* Vp = (const __bf16*)(smem + SM_KV(buf) + SM_VOFF);

      /* S[16x32] = Q(16x512) * K^T */
      v8f s0 = {}, s1 = {};
      for (int dt = 0; dt < 16; ++dt) {
        v16bf a;
        ((int4*)&a)[0] = *(const int4*)(qrow + dt * 32);
        ((int4*)&a)[1] = *(const int4*)(qrow + dt * 32 + 16);
        v16bf b0 = *(const v16bf*)(Kp + (dt * 2 + 0) * 512 + lane * 16);
        v16bf b1 = *(const v16bf*)(Kp + (dt * 2 + 1) * 512 + lane * 16);
        s0 = wmma_bf16(a, b0, s0);
        s1 = wmma_bf16(a, b1, s1);
      }

      /* scale + causal mask + row max */
      float rmax[8];
#pragma unroll
      for (int r = 0; r < 8; ++r) {
        int q = q0w + r + hi * 8;
        float v0 = s0[r] * scale, v1 = s1[r] * scale;
        if (j0 + row > q)      v0 = -__builtin_inff();
        if (j0 + 16 + row > q) v1 = -__builtin_inff();
        s0[r] = v0; s1[r] = v1;
        rmax[r] = fmaxf(v0, v1);
      }
#pragma unroll
      for (int r = 0; r < 8; ++r)
#pragma unroll
        for (int off = 1; off < 16; off <<= 1)
          rmax[r] = fmaxf(rmax[r], __shfl_xor(rmax[r], off, 16));

      float alpha[8], rsum[8];
#pragma unroll
      for (int r = 0; r < 8; ++r) {
        float mnew = fmaxf(m_i[r], rmax[r]);
        bool dead  = (mnew == -__builtin_inff());
        alpha[r]   = dead ? 1.0f : __expf(m_i[r] - mnew);
        float p0   = dead ? 0.0f : __expf(s0[r] - mnew);
        float p1   = dead ? 0.0f : __expf(s1[r] - mnew);
        m_i[r] = mnew;
        int m = r + hi * 8;
        Pw[m * 32 + row]      = (__bf16)p0;
        Pw[m * 32 + 16 + row] = (__bf16)p1;
        rsum[r] = p0 + p1;
      }
#pragma unroll
      for (int r = 0; r < 8; ++r) {
#pragma unroll
        for (int off = 1; off < 16; off <<= 1)
          rsum[r] += __shfl_xor(rsum[r], off, 16);
        l_i[r] = alpha[r] * l_i[r] + rsum[r];
      }

      __threadfence_block();
      /* P back as A-operand (16x32 bf16) */
      v16bf pA;
      ((int4*)&pA)[0] = *(const int4*)(Pw + row * 32 + hi * 8);
      ((int4*)&pA)[1] = *(const int4*)(Pw + row * 32 + hi * 8 + 16);

      /* O(16x512) += P * V, accumulator in LDS, fused rescale */
      for (int nt = 0; nt < 32; ++nt) {
        v16bf bv = *(const v16bf*)(Vp + nt * 512 + lane * 16);
        v8f c;
#pragma unroll
        for (int r = 0; r < 8; ++r) {
          int m = r + hi * 8;
          c[r] = Cw[m * 512 + nt * 16 + row] * alpha[r];
        }
        c = wmma_bf16(pA, bv, c);
#pragma unroll
        for (int r = 0; r < 8; ++r) {
          int m = r + hi * 8;
          Cw[m * 512 + nt * 16 + row] = c[r];
        }
      }
    }
    buf ^= 1;
  }

  /* epilogue: divide by l, write bf16 ctx */
  if (row == 0) {
#pragma unroll
    for (int r = 0; r < 8; ++r) Lw[hi * 8 + r] = l_i[r];
  }
  __threadfence_block();
  for (int e = lane; e < 16 * D_MODEL; e += 32) {
    int m = e >> 9, d = e & 511;
    ctx[(size_t)(tb + q0w + m) * D_MODEL + d] = (__bf16)(Cw[e] / Lw[m]);
  }
}

extern "C" void kernel_launch(void* const* d_in, const int* in_sizes, int n_in,
                              void* d_out, int out_size, void* d_ws, size_t ws_size,
                              hipStream_t stream) {
  (void)in_sizes; (void)n_in; (void)out_size; (void)ws_size;
  const float* x  = (const float*)d_in[0];
  const float* Wq = (const float*)d_in[1];
  const float* Wk = (const float*)d_in[2];
  const float* Wv = (const float*)d_in[3];
  const float* Wo = (const float*)d_in[4];

  char* ws = (char*)d_ws;
  __bf16* xb   = (__bf16*)(ws + OFF_XB);
  __bf16* Qb   = (__bf16*)(ws + OFF_QB);
  __bf16* Kb   = (__bf16*)(ws + OFF_KB);
  __bf16* Vt   = (__bf16*)(ws + OFF_VT);
  __bf16* ctxb = (__bf16*)(ws + OFF_CTX);
  __bf16* WqP  = (__bf16*)(ws + OFF_WQ);
  __bf16* WkP  = (__bf16*)(ws + OFF_WK);
  __bf16* WvP  = (__bf16*)(ws + OFF_WV);
  __bf16* WoP  = (__bf16*)(ws + OFF_WO);

  (void)hipFuncSetAttribute((const void*)k_attn,
                            hipFuncAttributeMaxDynamicSharedMemorySize, SM_TOTAL);

  const int NX = BT * D_MODEL;                     /* 8,388,608 */
  k_cast_bf16<<<4096, 256, 0, stream>>>(x, xb, NX);
  k_pack_w<<<1024, 256, 0, stream>>>(Wq, WqP);
  k_pack_w<<<1024, 256, 0, stream>>>(Wk, WkP);
  k_pack_w<<<1024, 256, 0, stream>>>(Wv, WvP);
  k_pack_w<<<1024, 256, 0, stream>>>(Wo, WoP);

  /* 256 row-groups x 32 col tiles = 8192 waves, 4 waves/block */
  k_gemm_qkv<<<2048, 128, 0, stream>>>(xb, WqP, WkP, WvP, Qb, Kb, Vt);

  dim3 ag(SEQ / 64, BATCH);
  k_attn<<<ag, 128, SM_TOTAL, stream>>>(Qb, Kb, Vt, ctxb);

  k_gemm_out<<<2048, 128, 0, stream>>>(ctxb, WoP, (float*)d_out);
}
